// PointMambaDGCNN_65962107732785
// MI455X (gfx1250) — compile-verified, matrix operand order checked by reading, not run on hardware
//
#include <hip/hip_runtime.h>
#include <math.h>

// ---------------- model constants ----------------
#define B_    4
#define N_    1024
#define KNN   20
#define DM_   1024
#define DI_   2048
#define DS_   16
#define DC_   4
#define DR_   64
#define L_    2048      // 2*N
#define BITS_ 10
#define GRID_ 0.02f

typedef __attribute__((ext_vector_type(16))) __bf16         v16bf;
typedef __attribute__((ext_vector_type(8)))  float          v8f;
typedef __attribute__((ext_vector_type(8)))  unsigned short us8;

union VFrag { v16bf v; us8 h[2]; };
union Pack4 { unsigned short u[4]; uint2 v; };

// float -> bf16 (RNE)
__device__ inline unsigned short f2bf_u(float f) {
  unsigned u = __builtin_bit_cast(unsigned, f);
  u += 0x7FFFu + ((u >> 16) & 1u);
  return (unsigned short)(u >> 16);
}
__device__ inline __bf16 f2bf(float f) {
  unsigned short h = f2bf_u(f);
  return __builtin_bit_cast(__bf16, h);
}

// K index inside a 32-wide chunk for element e (0..15) of the 16-bit A/B
// fragment (ISA 7.12.2).  Per lane the fragment is two contiguous 8-element
// runs: e in [0,8) -> K = 8*hl + e; e in [8,16) -> K = 16 + 8*hl + (e-8).
__device__ inline int wmma_k_of(int e, int hl) {
  return (e < 8) ? ((hl ? 8 : 0) + e) : (16 + (hl ? 8 : 0) + (e - 8));
}

__device__ inline v8f wmma_bf16(v16bf a, v16bf b, v8f c) {
  return __builtin_amdgcn_wmma_f32_16x16x32_bf16(false, a, false, b,
                                                 (short)0, c, false, false);
}

// load a 16-element W fragment (bf16 source, dense K) with 2x b128 loads
__device__ inline v16bf load_w_frag_fast(const unsigned short* __restrict__ r, int hl) {
  VFrag w;
  const unsigned short* r1 = r + (hl ? 8 : 0);
  w.h[0] = *(const us8*)(r1);
  w.h[1] = *(const us8*)(r1 + 16);
  return w.v;
}

// ---------------- generic WMMA GEMM:  out = epilogue(A(MxKd) * W(OxKd)^T) ---
// 4-wave workgroup computes a 16x256 tile.  Per K-chunk the block stages the
// 16x32 A sub-tile into LDS once (f32 -> bf16 during staging, boundary guards
// applied there); each wave reads its A fragment back as two ds_load_b128 and
// reuses it for 4 WMMAs against 4 weight fragments (16x64 columns per wave).
enum { EP_NONE = 0, EP_BIAS, EP_BIAS_GELU, EP_BN_LRELU, EP_BIAS_SOFTPLUS, EP_RES };

#define AS_PITCH 40   // ushorts per LDS row; 80B row stride keeps runs 16B-aligned

template <int EP>
__global__ __launch_bounds__(128)
void gemm_wmma(const float* __restrict__ A, int lda,
               const unsigned short* __restrict__ Wb,
               const float* __restrict__ p0, const float* __restrict__ p1,
               const float* __restrict__ p2, const float* __restrict__ p3,
               const float* __restrict__ res,
               float* __restrict__ out, int M, int Kd, int O) {
  __shared__ alignas(16) unsigned short As[16][AS_PITCH];
  int m0 = blockIdx.x * 16;
  int tid = threadIdx.x, wave = tid >> 5, lane = tid & 31;
  int hl = lane >> 4, q = lane & 15;
  int o0 = blockIdx.y * 256 + wave * 64;
  int nsub = (O - o0) >> 4;
  if (nsub > 4) nsub = 4;
  if (nsub < 0) nsub = 0;
  const unsigned short* Wr[4];
#pragma unroll
  for (int s = 0; s < 4; ++s)
    Wr[s] = Wb + (size_t)(o0 + s * 16 + q) * Kd;
  v8f acc[4] = {{}, {}, {}, {}};
  bool wfast = ((Kd & 31) == 0);
  // staging role: thread covers row sr, 4 consecutive K columns at sc0
  int sr = tid >> 3, sc0 = (tid & 7) * 4;
  int srow = m0 + sr;
  bool srv = srow < M;
  const float* Sp = A + (size_t)srow * lda;
  int nk = (Kd + 31) >> 5;
  for (int kc = 0; kc < nk; ++kc) {
    int kb = kc << 5;
    // ---- cooperative stage: 16x32 f32 -> bf16 in LDS ----
    Pack4 pk;
#pragma unroll
    for (int j = 0; j < 4; ++j) {
      int k = kb + sc0 + j;
      pk.u[j] = f2bf_u((srv && k < Kd) ? Sp[k] : 0.f);
    }
    *(uint2*)&As[sr][sc0] = pk.v;
    __syncthreads();
    // ---- consume: A fragment from LDS (2x ds_load_b128), 4 WMMAs ----
    VFrag af;
    af.h[0] = *(const us8*)&As[q][hl * 8];
    af.h[1] = *(const us8*)&As[q][16 + hl * 8];
#pragma unroll
    for (int s = 0; s < 4; ++s) {
      if (s >= nsub) break;
      v16bf w;
      if (wfast) {
        w = load_w_frag_fast(Wr[s] + kb, hl);
      } else {
#pragma unroll
        for (int e = 0; e < 16; ++e) {
          int k = kb + wmma_k_of(e, hl);
          unsigned short hv = (k < Kd) ? Wr[s][k] : (unsigned short)0;
          w[e] = __builtin_bit_cast(__bf16, hv);
        }
      }
      acc[s] = wmma_bf16(af.v, w, acc[s]);
    }
    __syncthreads();
  }
#pragma unroll
  for (int s = 0; s < 4; ++s) {
    if (s >= nsub) break;
    int n = o0 + s * 16 + q;
    float scale = 1.f, shift = 0.f;
    if (EP == EP_BIAS || EP == EP_BIAS_GELU || EP == EP_BIAS_SOFTPLUS) shift = p0[n];
    if (EP == EP_BN_LRELU) {
      float sc = p0[n] * rsqrtf(p3[n] + 1e-5f);
      scale = sc;
      shift = p1[n] - p2[n] * sc;
    }
#pragma unroll
    for (int r = 0; r < 8; ++r) {
      int m = m0 + r + (hl << 3);
      if (m >= M) continue;
      float v = acc[s][r];
      if (EP == EP_BIAS) v += shift;
      else if (EP == EP_BIAS_GELU) { v += shift; v = 0.5f * v * (1.f + erff(v * 0.70710678118f)); }
      else if (EP == EP_BN_LRELU)  { v = v * scale + shift; v = v >= 0.f ? v : 0.2f * v; }
      else if (EP == EP_BIAS_SOFTPLUS) { v += shift; v = v > 30.f ? v : log1pf(expf(v)); }
      else if (EP == EP_RES) v += res[(size_t)m * O + n];
      out[(size_t)m * O + n] = v;
    }
  }
}

// ---------------- fused EdgeConv: gather + WMMA + BN + LReLU + max over k ---
// 16 points x 64 output channels per wave; the gathered A fragment (the
// expensive part: neighbor-minus-center features) is reused for 4 WMMAs.
__global__ __launch_bounds__(32)
void edgeconv_wmma(const float* __restrict__ X, const int* __restrict__ knn,
                   const unsigned short* __restrict__ Wb,
                   const float* __restrict__ bw, const float* __restrict__ bb,
                   const float* __restrict__ bm, const float* __restrict__ bv,
                   float* __restrict__ out, int C, int O) {
  int b  = blockIdx.z;
  int n0 = blockIdx.x * 16, o0 = blockIdx.y * 64;
  int lane = threadIdx.x, hl = lane >> 4, q = lane & 15;
  int row = n0 + q;                       // A-matrix row (point index)
  const float* Xb  = X + (size_t)b * N_ * C;
  const float* ctr = Xb + (size_t)row * C;
  int twoC = 2 * C;
  bool fastw = ((twoC & 31) == 0);
  const unsigned short* Wr[4];
#pragma unroll
  for (int s = 0; s < 4; ++s)
    Wr[s] = Wb + (size_t)(o0 + s * 16 + q) * twoC;
  int nk = (twoC + 31) >> 5;
  float scl[4], shf[4];
#pragma unroll
  for (int s = 0; s < 4; ++s) {
    int n = o0 + s * 16 + q;
    float sc = bw[n] * rsqrtf(bv[n] + 1e-5f);
    scl[s] = sc;
    shf[s] = bb[n] - bm[n] * sc;
  }
  float mx[4][8];
#pragma unroll
  for (int s = 0; s < 4; ++s)
#pragma unroll
    for (int r = 0; r < 8; ++r) mx[s][r] = -3.3e38f;

  for (int kk = 0; kk < KNN; ++kk) {
    int nb = knn[((size_t)b * N_ + row) * KNN + kk];
    const float* nbp = Xb + (size_t)nb * C;
    v8f acc[4] = {{}, {}, {}, {}};
    for (int kc = 0; kc < nk; ++kc) {
      int kb2 = kc << 5;
      v16bf a;
#pragma unroll
      for (int e = 0; e < 16; ++e) {
        int k = kb2 + wmma_k_of(e, hl);
        float av = 0.f;
        if (k < twoC) av = (k < C) ? (nbp[k] - ctr[k]) : ctr[k - C];
        a[e] = f2bf(av);
      }
#pragma unroll
      for (int s = 0; s < 4; ++s) {
        v16bf w;
        if (fastw) {
          w = load_w_frag_fast(Wr[s] + kb2, hl);
        } else {
#pragma unroll
          for (int e = 0; e < 16; ++e) {
            int k = kb2 + wmma_k_of(e, hl);
            unsigned short hv = (k < twoC) ? Wr[s][k] : (unsigned short)0;
            w[e] = __builtin_bit_cast(__bf16, hv);
          }
        }
        acc[s] = wmma_bf16(a, w, acc[s]);
      }
    }
#pragma unroll
    for (int s = 0; s < 4; ++s)
#pragma unroll
      for (int r = 0; r < 8; ++r) {
        float v = acc[s][r] * scl[s] + shf[s];
        v = v >= 0.f ? v : 0.2f * v;
        mx[s][r] = fmaxf(mx[s][r], v);
      }
  }
#pragma unroll
  for (int s = 0; s < 4; ++s)
#pragma unroll
    for (int r = 0; r < 8; ++r) {
      int m = n0 + r + (hl << 3);
      out[((size_t)b * N_ + m) * O + o0 + s * 16 + q] = mx[s][r];
    }
}

// ---------------- misc small kernels ----------------
__global__ void cvt_bf16(const float* __restrict__ s, unsigned short* __restrict__ d, int n) {
  int i = blockIdx.x * blockDim.x + threadIdx.x;
  if (i < n) d[i] = f2bf_u(s[i]);
}

__global__ void transpose_pts(const float* __restrict__ x, float* __restrict__ pts) {
  int i = blockIdx.x * blockDim.x + threadIdx.x;
  if (i >= B_ * N_) return;
  int b = i / N_, n = i % N_;
#pragma unroll
  for (int a = 0; a < 3; ++a)
    pts[(size_t)i * 3 + a] = x[((size_t)b * 3 + a) * N_ + n];
}

__global__ void rownorm(const float* __restrict__ X, float* __restrict__ xx, int C) {
  int i = blockIdx.x * blockDim.x + threadIdx.x;
  if (i >= B_ * N_) return;
  const float* p = X + (size_t)i * C;
  float s = 0.f;
  for (int c = 0; c < C; ++c) s += p[c] * p[c];
  xx[i] = s;
}

// d[b,n,m] = 2*<x_n,x_m> - ||x_n||^2 - ||x_m||^2   (C <= 128)
__global__ __launch_bounds__(256)
void pairdist(const float* __restrict__ X, const float* __restrict__ xx,
              float* __restrict__ D, int C) {
  __shared__ float As[16][128];
  __shared__ float Bs[16][128];
  int b = blockIdx.z, n0 = blockIdx.x * 16, m0 = blockIdx.y * 16;
  int tx = threadIdx.x & 15, ty = threadIdx.x >> 4;
  const float* Xb = X + (size_t)b * N_ * C;
  for (int e = threadIdx.x; e < 16 * C; e += 256) {
    int r = e / C, c = e % C;
    As[r][c] = Xb[(size_t)(n0 + r) * C + c];
    Bs[r][c] = Xb[(size_t)(m0 + r) * C + c];
  }
  __syncthreads();
  float s = 0.f;
  for (int c = 0; c < C; ++c) s += As[ty][c] * Bs[tx][c];
  D[((size_t)b * N_ + n0 + ty) * N_ + m0 + tx] =
      2.f * s - xx[b * N_ + n0 + ty] - xx[b * N_ + m0 + tx];
}

// one wave per point: 20 rounds of argmax (ties -> lowest index, as lax.top_k)
__global__ __launch_bounds__(32)
void topk20(const float* __restrict__ D, int* __restrict__ idx) {
  int rowid = blockIdx.x;
  int lane = threadIdx.x;
  const float* d = D + (size_t)rowid * N_;
  float vals[32];
#pragma unroll
  for (int i = 0; i < 32; ++i) vals[i] = d[i * 32 + lane];
  unsigned taken = 0u;
  for (int r = 0; r < KNN; ++r) {
    float bvv = -3.4e38f;
    int bi = 0x7fffffff;
#pragma unroll
    for (int i = 0; i < 32; ++i) {
      if (!((taken >> i) & 1u)) {
        if (vals[i] > bvv) { bvv = vals[i]; bi = i * 32 + lane; }
      }
    }
    for (int off = 16; off > 0; off >>= 1) {
      float ov = __shfl_xor(bvv, off, 32);
      int   oi = __shfl_xor(bi, off, 32);
      if (ov > bvv || (ov == bvv && oi < bi)) { bvv = ov; bi = oi; }
    }
    if ((bi & 31) == lane) taken |= 1u << (bi >> 5);
    if (lane == 0) idx[(size_t)rowid * KNN + r] = bi;
  }
}

__global__ __launch_bounds__(256)
void concat4(const float* __restrict__ x1, const float* __restrict__ x2,
             const float* __restrict__ x3, const float* __restrict__ x4,
             float* __restrict__ out) {
  int i = blockIdx.x;
  float* o = out + (size_t)i * 512;
  for (int c = threadIdx.x; c < 512; c += 256) {
    float v;
    if      (c < 64)  v = x1[(size_t)i * 64  + c];
    else if (c < 128) v = x2[(size_t)i * 64  + (c - 64)];
    else if (c < 256) v = x3[(size_t)i * 128 + (c - 128)];
    else              v = x4[(size_t)i * 256 + (c - 256)];
    o[c] = v;
  }
}

__global__ __launch_bounds__(1024)
void gridmin(const float* __restrict__ pts, int* __restrict__ gmin) {
  int b = blockIdx.x, t = threadIdx.x;
  __shared__ int sm[1024];
  for (int a = 0; a < 3; ++a) {
    sm[t] = (int)floorf(pts[((size_t)b * N_ + t) * 3 + a] / GRID_);
    __syncthreads();
    for (int s = 512; s > 0; s >>= 1) {
      if (t < s) sm[t] = min(sm[t], sm[t + s]);
      __syncthreads();
    }
    if (t == 0) gmin[b * 3 + a] = sm[0];
    __syncthreads();
  }
}

// Skilling AxesToTranspose -> interleaved Hilbert code
__global__ void hilbert_code(const float* __restrict__ pts, const int* __restrict__ gmin,
                             int trans, int* __restrict__ code) {
  int i = blockIdx.x * blockDim.x + threadIdx.x;
  if (i >= B_ * N_) return;
  int b = i / N_;
  int g[3];
#pragma unroll
  for (int a = 0; a < 3; ++a) {
    int v = (int)floorf(pts[(size_t)i * 3 + a] / GRID_) - gmin[b * 3 + a];
    g[a] = min(max(v, 0), (1 << BITS_) - 1);
  }
  int xs[3];
  if (trans) { xs[0] = g[1]; xs[1] = g[0]; xs[2] = g[2]; }
  else       { xs[0] = g[0]; xs[1] = g[1]; xs[2] = g[2]; }
  for (int Q = 1 << (BITS_ - 1); Q > 1; Q >>= 1) {
    int P = Q - 1;
    if (xs[0] & Q) xs[0] ^= P;
    for (int j = 1; j < 3; ++j) {
      if (xs[j] & Q) {
        xs[0] ^= P;
      } else {
        int t = (xs[0] ^ xs[j]) & P;
        xs[0] ^= t;
        xs[j] ^= t;
      }
    }
  }
  xs[1] ^= xs[0];
  xs[2] ^= xs[1];
  int t2 = 0;
  for (int Q = 1 << (BITS_ - 1); Q > 1; Q >>= 1)
    if (xs[2] & Q) t2 ^= Q - 1;
  xs[0] ^= t2; xs[1] ^= t2; xs[2] ^= t2;
  int c = 0;
  for (int bit = BITS_ - 1; bit >= 0; --bit)
    for (int a = 0; a < 3; ++a) c = (c << 1) | ((xs[a] >> bit) & 1);
  code[i] = c;
}

// stable argsort of 1024 codes per batch via bitonic sort on (code,index)
__global__ __launch_bounds__(1024)
void bitonic_argsort(const int* __restrict__ code, int* __restrict__ order) {
  int b = blockIdx.x, t = threadIdx.x;
  __shared__ int sc[1024];
  __shared__ int si[1024];
  sc[t] = code[b * N_ + t];
  si[t] = t;
  __syncthreads();
  for (int k = 2; k <= 1024; k <<= 1)
    for (int j = k >> 1; j > 0; j >>= 1) {
      int ixj = t ^ j;
      if (ixj > t) {
        bool up = ((t & k) == 0);
        int c1 = sc[t], c2 = sc[ixj], i1 = si[t], i2 = si[ixj];
        bool gt = (c1 > c2) || (c1 == c2 && i1 > i2);
        if (gt == up) { sc[t] = c2; sc[ixj] = c1; si[t] = i2; si[ixj] = i1; }
      }
      __syncthreads();
    }
  order[b * N_ + t] = si[t];
}

__global__ __launch_bounds__(256)
void build_hs(const float* __restrict__ dg, const float* __restrict__ pos5,
              const int* __restrict__ of, const int* __restrict__ ob,
              const float* __restrict__ gamma, const float* __restrict__ beta,
              float* __restrict__ hs) {
  int b = blockIdx.y, l = blockIdx.x;
  int p = (l < N_) ? of[b * N_ + l] : ob[b * N_ + (l - N_)];
  const float* dgr = dg   + ((size_t)b * N_ + p) * DM_;
  const float* pr  = pos5 + ((size_t)b * N_ + p) * DM_;
  float* o = hs + ((size_t)b * L_ + l) * DM_;
  for (int c = threadIdx.x; c < DM_; c += 256)
    o[c] = dgr[c] * gamma[c] + beta[c] + pr[c];
}

__global__ __launch_bounds__(256)
void layernorm(const float* __restrict__ X, const float* __restrict__ w,
               const float* __restrict__ bsh, float* __restrict__ out, int Dch) {
  __shared__ float s1[256], s2[256];
  int row = blockIdx.x, t = threadIdx.x;
  const float* x = X + (size_t)row * Dch;
  float a = 0.f, b2 = 0.f;
  for (int c = t; c < Dch; c += 256) {
    float v = x[c];
    a += v;
    b2 += v * v;
  }
  s1[t] = a; s2[t] = b2;
  __syncthreads();
  for (int s = 128; s > 0; s >>= 1) {
    if (t < s) { s1[t] += s1[t + s]; s2[t] += s2[t + s]; }
    __syncthreads();
  }
  float mu = s1[0] / Dch;
  float var = s2[0] / Dch - mu * mu;
  float inv = rsqrtf(var + 1e-5f);
  float* o = out + (size_t)row * Dch;
  for (int c = t; c < Dch; c += 256) o[c] = (x[c] - mu) * inv * w[c] + bsh[c];
}

// causal depthwise conv (DC=4) + SiLU; xc = channels [0,DI) of xz
__global__ __launch_bounds__(256)
void dwconv_silu(const float* __restrict__ xz, const float* __restrict__ cw,
                 const float* __restrict__ cb, float* __restrict__ out) {
  int b = blockIdx.y, l = blockIdx.x;
  for (int d = threadIdx.x; d < DI_; d += 256) {
    float acc = cb[d];
#pragma unroll
    for (int j = 0; j < DC_; ++j) {
      int lp = l + j - (DC_ - 1);
      if (lp >= 0) acc += cw[d * DC_ + j] * xz[((size_t)b * L_ + lp) * (2 * DI_) + d];
    }
    out[((size_t)b * L_ + l) * DI_ + d] = acc / (1.f + expf(-acc));
  }
}

// selective scan: each thread owns one (b,d) with 16 states in registers;
// B_t / C_t broadcast through LDS; z-gating fused at the end of each step.
__global__ __launch_bounds__(256)
void ssm_scan(const float* __restrict__ dtb, const float* __restrict__ xconv,
              const float* __restrict__ xdbl, const float* __restrict__ xz,
              const float* __restrict__ A_log, const float* __restrict__ Dp,
              float* __restrict__ g) {
  int b = blockIdx.y;
  int d = blockIdx.x * 256 + threadIdx.x;
  int t = threadIdx.x;
  __shared__ float sB[DS_], sC[DS_];
  float A[DS_], h[DS_];
#pragma unroll
  for (int n = 0; n < DS_; ++n) {
    A[n] = -expf(A_log[(size_t)d * DS_ + n]);
    h[n] = 0.f;
  }
  float Dpd = Dp[d];
  for (int l = 0; l < L_; ++l) {
    size_t rb = (size_t)b * L_ + l;
    if (t < DS_)            sB[t]       = xdbl[rb * 96 + DR_ + t];
    else if (t < 2 * DS_)   sC[t - DS_] = xdbl[rb * 96 + DR_ + DS_ + (t - DS_)];
    __syncthreads();
    float dt = dtb[rb * DI_ + d];
    float u  = xconv[rb * DI_ + d];
    float du = dt * u;
    float y = 0.f;
#pragma unroll
    for (int n = 0; n < DS_; ++n) {
      h[n] = h[n] * expf(dt * A[n]) + du * sB[n];
      y += h[n] * sC[n];
    }
    float z = xz[rb * (2 * DI_) + DI_ + d];
    g[rb * DI_ + d] = (y + u * Dpd) * (z / (1.f + expf(-z)));
    __syncthreads();
  }
}

// 63 segments of mean+max pooling over x_all = concat(hsf[2N], dg[N])
__global__ __launch_bounds__(256)
void binpool(const float* __restrict__ hsf, const float* __restrict__ dg,
             float* __restrict__ bf) {
  int s = blockIdx.x, b = blockIdx.y;
  int nbin = 1, base = 0;
  while (s >= base + nbin) { base += nbin; nbin <<= 1; }
  int bin = s - base;
  int P = 3 * N_;
  int seglen = P / nbin, start = bin * seglen;
  for (int c = threadIdx.x; c < DM_; c += 256) {
    float sum = 0.f, mx = -3.4e38f;
    for (int r = 0; r < seglen; ++r) {
      int row = start + r;
      float v = (row < L_) ? hsf[((size_t)b * L_ + row) * DM_ + c]
                           : dg[((size_t)b * N_ + (row - L_)) * DM_ + c];
      sum += v;
      mx = fmaxf(mx, v);
    }
    bf[((size_t)s * B_ + b) * DM_ + c] = sum / (float)seglen + mx;
  }
}

// ---------------- host-side orchestration ----------------
extern "C" void kernel_launch(void* const* d_in, const int* in_sizes, int n_in,
                              void* d_out, int out_size, void* d_ws, size_t ws_size,
                              hipStream_t stream) {
  (void)in_sizes; (void)n_in; (void)out_size; (void)ws_size;
  const float* x      = (const float*)d_in[0];
  const float* c1_w   = (const float*)d_in[1];
  const float* c2_w   = (const float*)d_in[2];
  const float* c3_w   = (const float*)d_in[3];
  const float* c4_w   = (const float*)d_in[4];
  const float* c5_w   = (const float*)d_in[5];
  const float* bn1_w  = (const float*)d_in[6],  *bn1_b = (const float*)d_in[7];
  const float* bn1_m  = (const float*)d_in[8],  *bn1_v = (const float*)d_in[9];
  const float* bn2_w  = (const float*)d_in[10], *bn2_b = (const float*)d_in[11];
  const float* bn2_m  = (const float*)d_in[12], *bn2_v = (const float*)d_in[13];
  const float* bn3_w  = (const float*)d_in[14], *bn3_b = (const float*)d_in[15];
  const float* bn3_m  = (const float*)d_in[16], *bn3_v = (const float*)d_in[17];
  const float* bn4_w  = (const float*)d_in[18], *bn4_b = (const float*)d_in[19];
  const float* bn4_m  = (const float*)d_in[20], *bn4_v = (const float*)d_in[21];
  const float* bn5_w  = (const float*)d_in[22], *bn5_b = (const float*)d_in[23];
  const float* bn5_m  = (const float*)d_in[24], *bn5_v = (const float*)d_in[25];
  const float* pe1_w  = (const float*)d_in[26], *pe1_b = (const float*)d_in[27];
  const float* pe2_w  = (const float*)d_in[28], *pe2_b = (const float*)d_in[29];
  const float* gamma  = (const float*)d_in[30], *beta  = (const float*)d_in[31];
  const float* blkn_w = (const float*)d_in[32], *blkn_b= (const float*)d_in[33];
  const float* in_w   = (const float*)d_in[34];
  const float* conv_w = (const float*)d_in[35], *conv_b= (const float*)d_in[36];
  const float* xproj_w= (const float*)d_in[37];
  const float* dt_w   = (const float*)d_in[38], *dt_b  = (const float*)d_in[39];
  const float* A_log  = (const float*)d_in[40], *Dpp   = (const float*)d_in[41];
  const float* out_w  = (const float*)d_in[42];
  const float* normf_w= (const float*)d_in[43], *normf_b=(const float*)d_in[44];
  const float* ft_w   = (const float*)d_in[45], *ft_b  = (const float*)d_in[46];
  float* out = (float*)d_out;

  // workspace arena
  size_t off = 0;
  char* base = (char*)d_ws;
  auto alloc = [&](size_t nbytes) -> void* {
    void* p = base + off;
    off += (nbytes + 255) & ~(size_t)255;
    return p;
  };
  float* pts   = (float*)alloc((size_t)B_ * N_ * 3 * 4);
  float* xx    = (float*)alloc((size_t)B_ * N_ * 4);
  float* dist  = (float*)alloc((size_t)B_ * N_ * N_ * 4);
  int*   knn   = (int*)  alloc((size_t)B_ * N_ * KNN * 4);
  float* x1    = (float*)alloc((size_t)B_ * N_ * 64 * 4);
  float* x2    = (float*)alloc((size_t)B_ * N_ * 64 * 4);
  float* x3    = (float*)alloc((size_t)B_ * N_ * 128 * 4);
  float* x4    = (float*)alloc((size_t)B_ * N_ * 256 * 4);
  float* xcat  = (float*)alloc((size_t)B_ * N_ * 512 * 4);
  float* dg    = (float*)alloc((size_t)B_ * N_ * DM_ * 4);
  float* h1    = (float*)alloc((size_t)B_ * N_ * 512 * 4);
  float* pos5  = (float*)alloc((size_t)B_ * N_ * DM_ * 4);
  int*   gmin  = (int*)  alloc((size_t)B_ * 3 * 4);
  int*   codef = (int*)  alloc((size_t)B_ * N_ * 4);
  int*   codeb = (int*)  alloc((size_t)B_ * N_ * 4);
  int*   ordf  = (int*)  alloc((size_t)B_ * N_ * 4);
  int*   ordb  = (int*)  alloc((size_t)B_ * N_ * 4);
  float* hs    = (float*)alloc((size_t)B_ * L_ * DM_ * 4);
  float* lnb   = (float*)alloc((size_t)B_ * L_ * DM_ * 4);
  float* xz    = (float*)alloc((size_t)B_ * L_ * 2 * DI_ * 4);
  float* xconv = (float*)alloc((size_t)B_ * L_ * DI_ * 4);
  float* xdbl  = (float*)alloc((size_t)B_ * L_ * 96 * 4);
  float* dtb   = (float*)alloc((size_t)B_ * L_ * DI_ * 4);
  float* gbuf  = (float*)alloc((size_t)B_ * L_ * DI_ * 4);
  float* hs2   = (float*)alloc((size_t)B_ * L_ * DM_ * 4);
  float* hsf   = (float*)alloc((size_t)B_ * L_ * DM_ * 4);
  float* bfb   = (float*)alloc((size_t)63 * B_ * DM_ * 4);
  // bf16 weight copies (halve weight bandwidth; enable b128 fragment loads)
  unsigned short* c1b  = (unsigned short*)alloc((size_t)64 * 6 * 2);
  unsigned short* c2b  = (unsigned short*)alloc((size_t)64 * 128 * 2);
  unsigned short* c3b  = (unsigned short*)alloc((size_t)128 * 128 * 2);
  unsigned short* c4b  = (unsigned short*)alloc((size_t)256 * 256 * 2);
  unsigned short* c5b  = (unsigned short*)alloc((size_t)1024 * 512 * 2);
  unsigned short* pe1b = (unsigned short*)alloc((size_t)512 * 3 * 2);
  unsigned short* pe2b = (unsigned short*)alloc((size_t)1024 * 512 * 2);
  unsigned short* inwb = (unsigned short*)alloc((size_t)4096 * 1024 * 2);
  unsigned short* xpjb = (unsigned short*)alloc((size_t)96 * 2048 * 2);
  unsigned short* dtwb = (unsigned short*)alloc((size_t)2048 * 64 * 2);
  unsigned short* outwb= (unsigned short*)alloc((size_t)1024 * 2048 * 2);
  unsigned short* ftwb = (unsigned short*)alloc((size_t)256 * 1024 * 2);

  // convert all weights to bf16 once
  struct { const float* s; unsigned short* d; int n; } cvts[12] = {
    { c1_w, c1b, 64 * 6 },      { c2_w, c2b, 64 * 128 },
    { c3_w, c3b, 128 * 128 },   { c4_w, c4b, 256 * 256 },
    { c5_w, c5b, 1024 * 512 },  { pe1_w, pe1b, 512 * 3 },
    { pe2_w, pe2b, 1024 * 512 },{ in_w, inwb, 4096 * 1024 },
    { xproj_w, xpjb, 96 * 2048 },{ dt_w, dtwb, 2048 * 64 },
    { out_w, outwb, 1024 * 2048 },{ ft_w, ftwb, 256 * 1024 },
  };
  for (int i = 0; i < 12; ++i)
    cvt_bf16<<<(cvts[i].n + 255) / 256, 256, 0, stream>>>(cvts[i].s, cvts[i].d, cvts[i].n);

  int BN = B_ * N_;
  // (B,3,N) -> (B,N,3)
  transpose_pts<<<(BN + 255) / 256, 256, 0, stream>>>(x, pts);

  // ---- 4 EdgeConv stages (knn feature dims 3, 64, 64, 128) ----
  struct { const float* in; int C; int O; const unsigned short* w;
           const float *bw, *bb, *bm, *bv; float* o; } st[4] = {
    { pts, 3,   64,  c1b, bn1_w, bn1_b, bn1_m, bn1_v, x1 },
    { x1,  64,  64,  c2b, bn2_w, bn2_b, bn2_m, bn2_v, x2 },
    { x2,  64,  128, c3b, bn3_w, bn3_b, bn3_m, bn3_v, x3 },
    { x3,  128, 256, c4b, bn4_w, bn4_b, bn4_m, bn4_v, x4 },
  };
  for (int i = 0; i < 4; ++i) {
    rownorm<<<(BN + 255) / 256, 256, 0, stream>>>(st[i].in, xx, st[i].C);
    pairdist<<<dim3(N_ / 16, N_ / 16, B_), 256, 0, stream>>>(st[i].in, xx, dist, st[i].C);
    topk20<<<BN, 32, 0, stream>>>(dist, knn);
    edgeconv_wmma<<<dim3(N_ / 16, st[i].O / 64, B_), 32, 0, stream>>>(
        st[i].in, knn, st[i].w, st[i].bw, st[i].bb, st[i].bm, st[i].bv,
        st[i].o, st[i].C, st[i].O);
  }

  // dg = lrelu(bn5(concat(x1..x4) @ c5_w^T))
  concat4<<<BN, 256, 0, stream>>>(x1, x2, x3, x4, xcat);
  gemm_wmma<EP_BN_LRELU><<<dim3(BN / 16, 4), 128, 0, stream>>>(
      xcat, 512, c5b, bn5_w, bn5_b, bn5_m, bn5_v, nullptr, dg, BN, 512, DM_);

  // pos5 = gelu(pts @ pe1^T + b1) @ pe2^T + b2
  gemm_wmma<EP_BIAS_GELU><<<dim3(BN / 16, 2), 128, 0, stream>>>(
      pts, 3, pe1b, pe1_b, nullptr, nullptr, nullptr, nullptr, h1, BN, 3, 512);
  gemm_wmma<EP_BIAS><<<dim3(BN / 16, 4), 128, 0, stream>>>(
      h1, 512, pe2b, pe2_b, nullptr, nullptr, nullptr, nullptr, pos5, BN, 512, DM_);

  // Hilbert serialization orders (forward + transposed)
  gridmin<<<B_, 1024, 0, stream>>>(pts, gmin);
  hilbert_code<<<(BN + 255) / 256, 256, 0, stream>>>(pts, gmin, 0, codef);
  hilbert_code<<<(BN + 255) / 256, 256, 0, stream>>>(pts, gmin, 1, codeb);
  bitonic_argsort<<<B_, 1024, 0, stream>>>(codef, ordf);
  bitonic_argsort<<<B_, 1024, 0, stream>>>(codeb, ordb);

  // hs = [take(dg,of)*g+b ; take(dg,ob)*g+b] + [take(pos5,of) ; take(pos5,ob)]
  build_hs<<<dim3(L_, B_), 256, 0, stream>>>(dg, pos5, ordf, ordb, gamma, beta, hs);

  // ---- Mamba block ----
  int BL = B_ * L_;
  layernorm<<<BL, 256, 0, stream>>>(hs, blkn_w, blkn_b, lnb, DM_);
  gemm_wmma<EP_NONE><<<dim3(BL / 16, 16), 128, 0, stream>>>(
      lnb, DM_, inwb, nullptr, nullptr, nullptr, nullptr, nullptr, xz, BL, DM_, 2 * DI_);
  dwconv_silu<<<dim3(L_, B_), 256, 0, stream>>>(xz, conv_w, conv_b, xconv);
  gemm_wmma<EP_NONE><<<dim3(BL / 16, 1), 128, 0, stream>>>(
      xconv, DI_, xpjb, nullptr, nullptr, nullptr, nullptr, nullptr, xdbl, BL, DI_, 96);
  gemm_wmma<EP_BIAS_SOFTPLUS><<<dim3(BL / 16, 8), 128, 0, stream>>>(
      xdbl, 96, dtwb, dt_b, nullptr, nullptr, nullptr, nullptr, dtb, BL, DR_, DI_);
  ssm_scan<<<dim3(DI_ / 256, B_), 256, 0, stream>>>(dtb, xconv, xdbl, xz, A_log, Dpp, gbuf);
  gemm_wmma<EP_RES><<<dim3(BL / 16, 4), 128, 0, stream>>>(
      gbuf, DI_, outwb, nullptr, nullptr, nullptr, nullptr, hs, hs2, BL, DI_, DM_);

  // final LN, pooled bin features, head
  layernorm<<<BL, 256, 0, stream>>>(hs2, normf_w, normf_b, hsf, DM_);
  binpool<<<dim3(63, B_), 256, 0, stream>>>(hsf, dg, bfb);
  gemm_wmma<EP_BIAS><<<dim3((63 * B_ + 15) / 16, 1), 128, 0, stream>>>(
      bfb, DM_, ftwb, ft_b, nullptr, nullptr, nullptr, nullptr, out, 63 * B_, DM_, 256);
}
